// CausalSelfAttention_71871982731782
// MI455X (gfx1250) — matrix operand order repaired and checked by reading, not compiled, not run on hardware
//
#include <hip/hip_runtime.h>
#include <hip/hip_bf16.h>

// ---------------------------------------------------------------------------
// CDNA5 (gfx1250) causal self-attention, bf16 WMMA with fp32 accumulation.
// Pipeline: cast -> QKV GEMM (WMMA + TDM double-buffered LDS tiles)
//          -> flash attention (WMMA) -> proj GEMM (WMMA + TDM).
// ---------------------------------------------------------------------------

typedef __attribute__((ext_vector_type(16))) __bf16 v16bf;
typedef __attribute__((ext_vector_type(8)))  float  v8f;
typedef __attribute__((ext_vector_type(4)))  unsigned int v4u;
typedef __attribute__((ext_vector_type(8)))  int    v8i;
typedef __attribute__((ext_vector_type(4)))  int    v4i;

#define WMMA_BF16(a, b, c)                                                    \
  __builtin_amdgcn_wmma_f32_16x16x32_bf16(false, (a), false, (b), (short)0,   \
                                          (c), false, false)

#if __has_builtin(__builtin_amdgcn_tensor_load_to_lds)
#define USE_TDM 1
#if __has_include(<hip/amd_detail/amd_gfx1250_TDM.h>)
#define TDM_6ARG 1
#endif
#else
#define USE_TDM 0
#endif

// ---------------- Fragment loaders (contiguous k = 16*(lane/16) + s, wave32) --------

// A-matrix 16x32 bf16, row-major source with leading dim `ld`.
__device__ inline v16bf load_frag_a(const __bf16* __restrict__ p, int ld,
                                    int lane) {
  int row = lane & 15;
  int h8 = (lane >> 4) * 16;
  const __bf16* r = p + row * ld;
  v16bf a;
#pragma unroll
  for (int j = 0; j < 8; ++j) {
    int k = h8 + 2 * j;  // h*16+2*j
    a[2 * j] = r[k];
    a[2 * j + 1] = r[k + 1];
  }
  return a;
}

// B-matrix 32x16 bf16, row-major [K x N] source with leading dim `ld`.
__device__ inline v16bf load_frag_b(const __bf16* __restrict__ p, int ld,
                                    int lane) {
  int col = lane & 15;
  int kb = (lane >> 4) * 16;
  v16bf b;
#pragma unroll
  for (int j = 0; j < 8; ++j) {
    b[2 * j] = p[(kb + 2 * j) * ld + col];
    b[2 * j + 1] = p[(kb + 2 * j + 1) * ld + col];
  }
  return b;
}

// B-matrix 32x16 where logical B[k][n] = src[n*ld + k]  (K^T access).
__device__ inline v16bf load_frag_bT(const __bf16* __restrict__ p, int ld,
                                     int lane) {
  int col = lane & 15;
  int kb = (lane >> 4) * 16;
  const __bf16* r = p + col * ld + kb;
  v16bf b;
#pragma unroll
  for (int j = 0; j < 16; ++j) b[j] = r[j];
  return b;
}

__device__ inline void copy32B(__bf16* dst, const __bf16* src) {
  const uint4* s = reinterpret_cast<const uint4*>(src);
  uint4* d = reinterpret_cast<uint4*>(dst);
  d[0] = s[0];
  d[1] = s[1];
}

// ---------------- TDM descriptor builders (D# per ISA ch.8 / 08_async) -----
#if USE_TDM
// Group 0: count=1 (word0), lds_addr (word1), global_addr[56:0] + type=2.
__device__ inline v4u tdm_group0(unsigned lds_off, unsigned long long ga) {
  v4u g;
  g[0] = 1u;  // count=1, is_restore=0, gather_mode=0
  g[1] = lds_off;
  g[2] = (unsigned)ga;
  g[3] = ((unsigned)(ga >> 32) & 0x01FFFFFFu) | 0x80000000u;  // type=2
  return g;
}

// Group 1: 2-byte elements, 2-D tile, LDS pad enabled.
// pad_interval: row bytes = 2^(pi+3); pad_amount: (pa+1) DWORDs appended.
__device__ inline v8i tdm_group1(unsigned tdim0, unsigned tdim1,
                                 unsigned tile0, unsigned tile1,
                                 unsigned long long stride0, unsigned pi,
                                 unsigned pa) {
  v8i g;
  unsigned w0 = (1u << 16)      // data_size = 1 -> 2 bytes
              | (1u << 20)      // pad_enable
              | (pi << 22)      // pad_interval
              | (pa << 25);     // pad_amount
  g[0] = (int)w0;
  g[1] = (int)((tdim0 & 0xFFFFu) << 16);                    // dim0[15:0]
  g[2] = (int)((tdim0 >> 16) | ((tdim1 & 0xFFFFu) << 16));  // dim0 hi, dim1 lo
  g[3] = (int)((tdim1 >> 16) | (tile0 << 16));              // dim1 hi, tile0
  g[4] = (int)tile1;                                        // tile1, tile2=0
  g[5] = (int)(unsigned)(stride0 & 0xFFFFFFFFu);
  g[6] = (int)(unsigned)((stride0 >> 32) & 0xFFFFu);        // dim1_stride=0
  g[7] = 0;
  return g;
}

__device__ inline void tdm_load(unsigned lds_off, const __bf16* gptr,
                                unsigned tdim0, unsigned tdim1, unsigned tile0,
                                unsigned tile1, unsigned long long stride0,
                                unsigned pi, unsigned pa) {
  v4u g0 = tdm_group0(lds_off, (unsigned long long)(size_t)(const void*)gptr);
  v8i g1 = tdm_group1(tdim0, tdim1, tile0, tile1, stride0, pi, pa);
  v4i z4 = {};
#if defined(TDM_6ARG)
  v8i z8 = {};
  __builtin_amdgcn_tensor_load_to_lds(g0, g1, z4, z4, z8, 0);
#else
  __builtin_amdgcn_tensor_load_to_lds(g0, g1, z4, z4, 0);
#endif
}
#endif  // USE_TDM

// ---------------- cast fp32 -> bf16 ----------------------------------------
__global__ void cast_f32_to_bf16(const float* __restrict__ src,
                                 __bf16* __restrict__ dst, int n) {
  int i = blockIdx.x * blockDim.x + threadIdx.x;
  int stride = gridDim.x * blockDim.x;
  for (; i < n; i += stride) dst[i] = (__bf16)src[i];
}

// ---------------- tiled WMMA GEMM ------------------------------------------
// C[M,N] = A[M,K] * B[K,N] + bias.  M=8192, K=768.  Tile 128x128x32.
// 256 threads = 8 waves; wave (wm,wn) computes 64x32 via 4x2 WMMA frags.
// Tiles staged global->LDS by the Tensor Data Mover (wave 0 issues the
// descriptors; TENSORcnt double-buffering overlaps DMA with WMMA).
// MODE 0: scatter bf16 into q/k/v [B,H,T,D].  MODE 1: fp32 row-major out.
template <int N, int MODE>
__global__ __launch_bounds__(256) void gemm_bf16_wmma(
    const __bf16* __restrict__ A, const __bf16* __restrict__ B,
    const float* __restrict__ bias, __bf16* __restrict__ qb,
    __bf16* __restrict__ kb, __bf16* __restrict__ vb,
    float* __restrict__ outf) {
  constexpr int K = 768;
  constexpr int LDA = 40;   // 32 + 8 pad (TDM: pi=3 -> 64B rows, pa=3 -> +16B)
  constexpr int LDB = 136;  // 128 + 8 pad (TDM: pi=5 -> 256B rows, pa=3)
  __shared__ __attribute__((aligned(16))) __bf16 sA[2][128 * LDA];
  __shared__ __attribute__((aligned(16))) __bf16 sB[2][32 * LDB];

  int tid = threadIdx.x;
  int lane = tid & 31;
  int wave = tid >> 5;
  int wm = (wave >> 2) * 64;  // 0 or 64
  int wn = (wave & 3) * 32;   // 0..96
  int m0 = blockIdx.y * 128;
  int n0 = blockIdx.x * 128;

  v8f acc[4][2];
#pragma unroll
  for (int i = 0; i < 4; ++i)
#pragma unroll
    for (int j = 0; j < 2; ++j) acc[i][j] = {};

#if USE_TDM
  // Prologue: DMA first k-step into buffer 0.
  if (wave == 0) {
    tdm_load((unsigned)(size_t)(void*)&sA[0][0], &A[(size_t)m0 * K], K, 8192,
             32, 128, (unsigned long long)K, 3, 3);
    tdm_load((unsigned)(size_t)(void*)&sB[0][0], &B[n0], N, K, 128, 32,
             (unsigned long long)N, 5, 3);
  }
  int buf = 0;
  for (int k0 = 0; k0 < K; k0 += 32) {
    __syncthreads();  // everyone done reading buf^1 from previous iteration
    bool more = (k0 + 32) < K;
    if (wave == 0) {
      if (more) {
        tdm_load((unsigned)(size_t)(void*)&sA[buf ^ 1][0],
                 &A[(size_t)m0 * K + k0 + 32], K, 8192, 32, 128,
                 (unsigned long long)K, 3, 3);
        tdm_load((unsigned)(size_t)(void*)&sB[buf ^ 1][0],
                 &B[(size_t)(k0 + 32) * N + n0], N, K, 128, 32,
                 (unsigned long long)N, 5, 3);
        // leave only the prefetch pair outstanding -> current buf complete
        __builtin_amdgcn_s_wait_tensorcnt(2);
      } else {
        __builtin_amdgcn_s_wait_tensorcnt(0);
      }
    }
    __syncthreads();  // buf is populated for all waves

    v16bf af[4];
#pragma unroll
    for (int i = 0; i < 4; ++i)
      af[i] = load_frag_a(&sA[buf][(wm + i * 16) * LDA], LDA, lane);
    v16bf bfr[2];
#pragma unroll
    for (int j = 0; j < 2; ++j)
      bfr[j] = load_frag_b(&sB[buf][wn + j * 16], LDB, lane);
#pragma unroll
    for (int i = 0; i < 4; ++i)
#pragma unroll
      for (int j = 0; j < 2; ++j)
        acc[i][j] = WMMA_BF16(af[i], bfr[j], acc[i][j]);
    buf ^= 1;
  }
#else
  // Fallback: vector-load staging (single buffer).
  int tra = tid >> 1, sega = (tid & 1) * 16;  // A: 128 rows x 32
  int trb = tid >> 3, segb = (tid & 7) * 16;  // B: 32 rows x 128
  for (int k0 = 0; k0 < K; k0 += 32) {
    __syncthreads();
    copy32B(&sA[0][tra * LDA + sega], &A[(size_t)(m0 + tra) * K + k0 + sega]);
    copy32B(&sB[0][trb * LDB + segb], &B[(size_t)(k0 + trb) * N + n0 + segb]);
    __syncthreads();
    v16bf af[4];
#pragma unroll
    for (int i = 0; i < 4; ++i)
      af[i] = load_frag_a(&sA[0][(wm + i * 16) * LDA], LDA, lane);
    v16bf bfr[2];
#pragma unroll
    for (int j = 0; j < 2; ++j)
      bfr[j] = load_frag_b(&sB[0][wn + j * 16], LDB, lane);
#pragma unroll
    for (int i = 0; i < 4; ++i)
#pragma unroll
      for (int j = 0; j < 2; ++j)
        acc[i][j] = WMMA_BF16(af[i], bfr[j], acc[i][j]);
  }
#endif

  int cl = lane & 15;
  int hf = lane >> 4;
#pragma unroll
  for (int i = 0; i < 4; ++i) {
#pragma unroll
    for (int j = 0; j < 2; ++j) {
#pragma unroll
      for (int e = 0; e < 8; ++e) {
        int row = m0 + wm + i * 16 + e + 8 * hf;
        int col = n0 + wn + j * 16 + cl;
        float val = acc[i][j][e] + bias[col];
        if (MODE == 0) {
          // qkv column: which*768 + head*64 + d  -> [B,H,T,D]
          int which = col / 768;
          int rem = col - which * 768;
          int hd = rem >> 6;
          int dd = rem & 63;
          int bi = row >> 10;
          int tt = row & 1023;
          __bf16* dst = (which == 0) ? qb : ((which == 1) ? kb : vb);
          dst[((((size_t)bi * 12 + hd) << 10) + tt) * 64 + dd] = (__bf16)val;
        } else {
          outf[(size_t)row * N + col] = val;
        }
      }
    }
  }
}

// ---------------- flash attention ------------------------------------------
// grid = (B*H, T/64), block = 128 (4 waves).  Wave owns 16 queries.
// S = Q K^T (2 k-steps of 32 over D=64), online softmax fp32,
// P staged to LDS as bf16 (C-frag -> A-frag relayout), O += P V.
__global__ __launch_bounds__(128) void attn_flash_wmma(
    const __bf16* __restrict__ Q, const __bf16* __restrict__ Kt,
    const __bf16* __restrict__ V, __bf16* __restrict__ Y) {
  constexpr int T = 1024, D = 64, H = 12, LDP = 40;
  __shared__ __bf16 sP[4][16 * LDP];

  int lane = threadIdx.x & 31;
  int wave = threadIdx.x >> 5;
  int bh = blockIdx.x;
  int b = bh / H;
  int h = bh % H;
  size_t base = (size_t)bh * T * D;
  const __bf16* Qp = Q + base;
  const __bf16* Kp = Kt + base;
  const __bf16* Vp = V + base;
  int q0 = blockIdx.y * 64 + wave * 16;

  // Loop-invariant Q fragments (16 queries x D=64 -> two K=32 A-frags)
  v16bf aq0 = load_frag_a(Qp + (size_t)q0 * D, D, lane);
  v16bf aq1 = load_frag_a(Qp + (size_t)q0 * D + 32, D, lane);

  v8f o[4];
#pragma unroll
  for (int i = 0; i < 4; ++i) o[i] = {};
  float mrow[8], lrow[8];
#pragma unroll
  for (int e = 0; e < 8; ++e) {
    mrow[e] = -__builtin_inff();
    lrow[e] = 0.0f;
  }

  int cl = lane & 15;
  int hf = lane >> 4;
  __bf16* pw = sP[wave];
  const float scale = 0.125f;  // 1/sqrt(64)
  int nkb = (q0 + 15) / 32 + 1;

  for (int kbk = 0; kbk < nkb; ++kbk) {
    int kbase = kbk * 32;
    // S = Q K^T : keys kbase..kbase+31 split into two 16-col frags
    v16bf bk00 = load_frag_bT(Kp + (size_t)kbase * D, D, lane);       // d 0-31
    v16bf bk01 = load_frag_bT(Kp + (size_t)kbase * D + 32, D, lane);  // d 32-63
    v16bf bk10 = load_frag_bT(Kp + (size_t)(kbase + 16) * D, D, lane);
    v16bf bk11 = load_frag_bT(Kp + (size_t)(kbase + 16) * D + 32, D, lane);
    v8f s0 = {}, s1 = {};
    s0 = WMMA_BF16(aq0, bk00, s0);
    s0 = WMMA_BF16(aq1, bk01, s0);
    s1 = WMMA_BF16(aq0, bk10, s1);
    s1 = WMMA_BF16(aq1, bk11, s1);

#pragma unroll
    for (int e = 0; e < 8; ++e) {
      int qg = q0 + e + 8 * hf;
      float v0 = s0[e] * scale;
      float v1 = s1[e] * scale;
      if (kbase + cl > qg) v0 = -__builtin_inff();
      if (kbase + 16 + cl > qg) v1 = -__builtin_inff();
      // row max across the 16-lane half (row lives in one VGPR slot/half)
      float mx = fmaxf(fmaxf(v0, v1), mrow[e]);
      mx = fmaxf(mx, __shfl_xor(mx, 1, 32));
      mx = fmaxf(mx, __shfl_xor(mx, 2, 32));
      mx = fmaxf(mx, __shfl_xor(mx, 4, 32));
      mx = fmaxf(mx, __shfl_xor(mx, 8, 32));
      float alpha = __expf(mrow[e] - mx);  // first block: exp(-inf)=0
      float p0 = __expf(v0 - mx);
      float p1 = __expf(v1 - mx);
      float rs = p0 + p1;
      rs += __shfl_xor(rs, 1, 32);
      rs += __shfl_xor(rs, 2, 32);
      rs += __shfl_xor(rs, 4, 32);
      rs += __shfl_xor(rs, 8, 32);
      lrow[e] = lrow[e] * alpha + rs;
      mrow[e] = mx;
#pragma unroll
      for (int ni = 0; ni < 4; ++ni) o[ni][e] *= alpha;
      pw[(e + 8 * hf) * LDP + cl] = (__bf16)p0;
      pw[(e + 8 * hf) * LDP + cl + 16] = (__bf16)p1;
    }
    // wave-local LDS RAW: per-wave DS ops are in-order; explicit wait anyway
    asm volatile("s_wait_dscnt 0x0" ::: "memory");
    v16bf ap = load_frag_a(pw, LDP, lane);
#pragma unroll
    for (int ni = 0; ni < 4; ++ni) {
      v16bf bv = load_frag_b(Vp + (size_t)kbase * D + ni * 16, D, lane);
      o[ni] = WMMA_BF16(ap, bv, o[ni]);
    }
  }

  // O /= l, write into y [B,T,C] (c = h*64+d) as bf16 for the proj GEMM
#pragma unroll
  for (int ni = 0; ni < 4; ++ni) {
#pragma unroll
    for (int e = 0; e < 8; ++e) {
      int t = q0 + e + 8 * hf;
      int c = h * 64 + ni * 16 + cl;
      Y[((size_t)b * T + t) * 768 + c] = (__bf16)(o[ni][e] / lrow[e]);
    }
  }
}

// ---------------- host side ------------------------------------------------
extern "C" void kernel_launch(void* const* d_in, const int* in_sizes, int n_in,
                              void* d_out, int out_size, void* d_ws,
                              size_t ws_size, hipStream_t stream) {
  constexpr int B = 8, T = 1024, C = 768, H = 12, D = 64;
  constexpr int BT = B * T;             // 8192
  constexpr int N3 = 3 * C;             // 2304
  constexpr size_t XN = (size_t)BT * C;         // 6291456
  constexpr size_t WA = (size_t)C * N3;         // 1769472
  constexpr size_t WP = (size_t)C * C;          // 589824
  constexpr size_t QN = (size_t)B * H * T * D;  // 6291456

  const float* x = (const float*)d_in[0];
  const float* Wattn = (const float*)d_in[1];
  const float* battn = (const float*)d_in[2];
  const float* Wproj = (const float*)d_in[3];
  const float* bproj = (const float*)d_in[4];
  float* out = (float*)d_out;

  __bf16* xb = (__bf16*)d_ws;
  __bf16* wab = xb + XN;
  __bf16* wpb = wab + WA;
  __bf16* qb = wpb + WP;
  __bf16* kb = qb + QN;
  __bf16* vb = kb + QN;
  __bf16* yb = vb + QN;

  auto blocks = [](size_t n) { return (unsigned)((n + 255) / 256); };
  cast_f32_to_bf16<<<blocks(XN), 256, 0, stream>>>(x, xb, (int)XN);
  cast_f32_to_bf16<<<blocks(WA), 256, 0, stream>>>(Wattn, wab, (int)WA);
  cast_f32_to_bf16<<<blocks(WP), 256, 0, stream>>>(Wproj, wpb, (int)WP);

  // QKV: [8192,768] x [768,2304] -> scatter q/k/v [B,H,T,D] bf16
  gemm_bf16_wmma<N3, 0><<<dim3(N3 / 128, BT / 128), 256, 0, stream>>>(
      xb, wab, battn, qb, kb, vb, nullptr);

  // flash attention -> y [B,T,C] bf16
  attn_flash_wmma<<<dim3(B * H, T / 64), 128, 0, stream>>>(qb, kb, vb, yb);

  // proj: [8192,768] x [768,768] + bias -> fp32 out
  gemm_bf16_wmma<C, 1><<<dim3(C / 128, BT / 128), 256, 0, stream>>>(
      yb, wpb, bproj, nullptr, nullptr, nullptr, out);
}
